// Qwen3VLAttention_57475252355438
// MI455X (gfx1250) — matrix operand order repaired and checked by reading, not compiled, not run on hardware
//
#include <hip/hip_runtime.h>

#define S_LEN 2048
#define HID   2048
#define NHQ   16
#define NKV   8
#define HD    128
#define ATT_SCALE 0.08838834764831845f  // 128^-0.5
#define NEG_BIG  -1.0e30f

typedef __attribute__((ext_vector_type(16))) __bf16 v16bf;
typedef __attribute__((ext_vector_type(8)))  float  v8f;
typedef __attribute__((ext_vector_type(8)))  unsigned int v8u;

union FragAB { v16bf bf; v8u u; };

__device__ __forceinline__ unsigned short f2bf(float f) {
  unsigned u = __float_as_uint(f);
  u += 0x7FFFu + ((u >> 16) & 1u);
  return (unsigned short)(u >> 16);
}

// ---- CDNA5 async memory->LDS copy (16B per lane), tracked by ASYNCcnt ----
__device__ __forceinline__ void async_ld_b128(void* lds_dst, const void* gsrc) {
  asm volatile("global_load_async_to_lds_b128 %0, %1, off"
               :: "v"((unsigned)(unsigned long long)lds_dst),
                  "v"((unsigned long long)gsrc)
               : "memory");
}
__device__ __forceinline__ void wait_async0() {
  asm volatile("s_wait_asynccnt 0x0" ::: "memory");
}
__device__ __forceinline__ void wait_async4() {
  asm volatile("s_wait_asynccnt 0x4" ::: "memory");
}
__device__ __forceinline__ void wait_async6() {
  asm volatile("s_wait_asynccnt 0x6" ::: "memory");
}

// ---------------- conversion kernels ----------------
__global__ __launch_bounds__(256) void cvt_bf16(const float* __restrict__ x,
                                                unsigned short* __restrict__ y, int n) {
  int i = blockIdx.x * 256 + threadIdx.x;
  if (i < n) y[i] = f2bf(x[i]);
}

// pack transposed [wq | wk | wv] -> W^T [4096 n][2048 k]
__global__ __launch_bounds__(256) void pack_w_t(const float* __restrict__ wq,
                                                const float* __restrict__ wk,
                                                const float* __restrict__ wv,
                                                unsigned short* __restrict__ w) {
  int i = blockIdx.x * 256 + threadIdx.x;          // n*2048 + k
  int n = i >> 11;
  int k = i & 2047;
  float v;
  if (n < 2048)      v = wq[k * 2048 + n];
  else if (n < 3072) v = wk[k * 1024 + (n - 2048)];
  else               v = wv[k * 1024 + (n - 3072)];
  w[i] = f2bf(v);
}

// transpose-convert square 2048x2048: dst[n][k] = src[k][n]
__global__ __launch_bounds__(256) void cvt_bf16_t(const float* __restrict__ src,
                                                  unsigned short* __restrict__ dst) {
  int i = blockIdx.x * 256 + threadIdx.x;
  int n = i >> 11;
  int k = i & 2047;
  dst[i] = f2bf(src[(size_t)k * 2048 + n]);
}

// ---------------- bf16 WMMA GEMM: C[f32 MxN] = A[bf16 MxK] * Bt[bf16 NxK] ----------------
// block tile 128x256, 8 waves of 64x64 (16 WMMAs / wave / k-step),
// async double-buffered LDS pipeline
__global__ __launch_bounds__(256) void gemm_bf16(const unsigned short* __restrict__ A,
                                                 const unsigned short* __restrict__ Bt,
                                                 float* __restrict__ C,
                                                 int M, int N, int K) {
  __shared__ unsigned short As[2][128 * 40];   // [row][k], pitch 40 (80B, 16B-aligned)
  __shared__ unsigned short Bs[2][256 * 40];   // [col][k], pitch 40

  const int nb   = N >> 8;
  const int bx   = blockIdx.x % nb;
  const int by   = blockIdx.x / nb;
  const int tid  = threadIdx.x;
  const int wave = tid >> 5;
  const int lane = tid & 31;
  const int hi   = lane >> 4;
  const int ln   = lane & 15;
  const int wm   = (wave >> 2) * 64;   // 0 / 64
  const int wn   = (wave & 3)  * 64;   // 0..192

  const int arow  = tid >> 1;          // 0..127
  const int ahalf = tid & 1;
  const unsigned short* gA = A  + (size_t)(by * 128 + arow) * K + ahalf * 16;
  const int aofs = arow * 40 + ahalf * 16;
  const unsigned short* gB = Bt + (size_t)(bx * 256 + tid) * K;   // one B row (C col) per thread
  const int bofs = tid * 40;

  auto stage = [&](int buf, int k0) {
    async_ld_b128(&As[buf][aofs],      gA + k0);
    async_ld_b128(&As[buf][aofs + 8],  gA + k0 + 8);
    async_ld_b128(&Bs[buf][bofs],      gB + k0);
    async_ld_b128(&Bs[buf][bofs + 8],  gB + k0 + 8);
    async_ld_b128(&Bs[buf][bofs + 16], gB + k0 + 16);
    async_ld_b128(&Bs[buf][bofs + 24], gB + k0 + 24);
  };

  v8f acc[4][4] = {};

  auto compute = [&](int cur) {
    FragAB a[4], b[4];
    #pragma unroll
    for (int i = 0; i < 4; ++i) {
      int r = wm + i * 16 + ln;
      #pragma unroll
      for (int p = 0; p < 8; ++p) {
        int kb = (p < 4 ? 2 * p : 2 * p + 8) + 8 * hi;
        a[i].u[p] = *(const unsigned*)(&As[cur][r * 40 + kb]);
      }
    }
    #pragma unroll
    for (int j = 0; j < 4; ++j) {
      int c = wn + j * 16 + ln;
      #pragma unroll
      for (int p = 0; p < 8; ++p) {
        int kb = (p < 4 ? 2 * p : 2 * p + 8) + 8 * hi;
        b[j].u[p] = *(const unsigned*)(&Bs[cur][c * 40 + kb]);
      }
    }
    #pragma unroll
    for (int i = 0; i < 4; ++i)
      #pragma unroll
      for (int j = 0; j < 4; ++j)
        acc[i][j] = __builtin_amdgcn_wmma_f32_16x16x32_bf16(
            false, a[i].bf, false, b[j].bf, (short)0, acc[i][j], false, false);
  };

  stage(0, 0);
  int k0 = 0;
  for (; k0 + 32 < K; k0 += 32) {           // steady state: prefetch + wait<=6
    const int cur = (k0 >> 5) & 1;
    stage(cur ^ 1, k0 + 32);
    wait_async6();
    __syncthreads();
    compute(cur);
    __syncthreads();
  }
  wait_async0();                            // last tile
  __syncthreads();
  compute((k0 >> 5) & 1);

  #pragma unroll
  for (int i = 0; i < 4; ++i)
    #pragma unroll
    for (int j = 0; j < 4; ++j) {
      int col = bx * 256 + wn + j * 16 + ln;
      #pragma unroll
      for (int r = 0; r < 8; ++r) {
        int rr = by * 128 + wm + i * 16 + r + 8 * hi;
        C[(size_t)rr * N + col] = acc[i][j][r];
      }
    }
}

// ---------------- RMSNorm + RoPE + layout shuffle ----------------
// qkv f32 [S][4096]; outputs: q_bf [S][NHQ*HD], k_bf [NKV][S][HD], v_t [NKV][HD][S]
__global__ __launch_bounds__(256) void norm_rope(const float* __restrict__ qkv,
                                                 const float* __restrict__ cosb,
                                                 const float* __restrict__ sinb,
                                                 const float* __restrict__ qw,
                                                 const float* __restrict__ kw,
                                                 unsigned short* __restrict__ q_bf,
                                                 unsigned short* __restrict__ k_bf,
                                                 unsigned short* __restrict__ v_t) {
  int gw   = (blockIdx.x * 256 + threadIdx.x) >> 5;
  int lane = threadIdx.x & 31;
  int s    = gw >> 5;
  int slot = gw & 31;        // 0-15: q head, 16-23: k head, 24-31: v head
  int col;
  if (slot < 16)      col = slot * HD;
  else if (slot < 24) col = 2048 + (slot - 16) * HD;
  else                col = 3072 + (slot - 24) * HD;
  const float* x = qkv + (size_t)s * 4096 + col;
  int d = lane * 4;
  float xv[4];
  #pragma unroll
  for (int t = 0; t < 4; ++t) xv[t] = x[d + t];

  if (slot < 24) {
    float ss = xv[0] * xv[0] + xv[1] * xv[1] + xv[2] * xv[2] + xv[3] * xv[3];
    #pragma unroll
    for (int m = 16; m >= 1; m >>= 1) ss += __shfl_xor(ss, m, 32);
    float inv = rsqrtf(ss * (1.0f / HD) + 1e-6f);
    const float* w = (slot < 16) ? qw : kw;
    float xn[4];
    #pragma unroll
    for (int t = 0; t < 4; ++t) xn[t] = xv[t] * inv * w[d + t];
    float pn[4];
    #pragma unroll
    for (int t = 0; t < 4; ++t) pn[t] = __shfl_xor(xn[t], 16, 32);  // partner d +/- 64
    bool lo = (d < 64);
    float o[4];
    #pragma unroll
    for (int t = 0; t < 4; ++t) {
      float rot = lo ? -pn[t] : pn[t];
      o[t] = xn[t] * cosb[s * HD + d + t] + rot * sinb[s * HD + d + t];
    }
    if (slot < 16) {
      unsigned short* dst = q_bf + (size_t)s * (NHQ * HD) + slot * HD + d;
      #pragma unroll
      for (int t = 0; t < 4; ++t) dst[t] = f2bf(o[t]);
    } else {
      int h = slot - 16;
      unsigned short* dst = k_bf + ((size_t)h * S_LEN + s) * HD + d;
      #pragma unroll
      for (int t = 0; t < 4; ++t) dst[t] = f2bf(o[t]);
    }
  } else {
    int h = slot - 24;
    #pragma unroll
    for (int t = 0; t < 4; ++t)
      v_t[((size_t)h * HD + d + t) * S_LEN + s] = f2bf(xv[t]);  // transposed [h][d][s]
  }
}

// ---------------- flash attention: one block = (head, 128 q rows), 8 waves x 16 rows ----------------
__global__ __launch_bounds__(256) void attn(const unsigned short* __restrict__ q_bf,
                                            const unsigned short* __restrict__ k_bf,
                                            const unsigned short* __restrict__ v_t,
                                            unsigned short* __restrict__ o_bf) {
  __shared__ unsigned short Ks[2][32 * 136];   // [key][d], pitch 136 (272B, 16B-aligned)
  __shared__ unsigned short Vs[2][128 * 40];   // [d][key], pitch 40
  __shared__ unsigned short Ps[8][16 * 40];    // per-wave P tile [row][key]

  const int h    = blockIdx.x >> 4;
  const int qb   = blockIdx.x & 15;
  const int hkv  = h >> 1;
  const int tid  = threadIdx.x;
  const int wave = tid >> 5;
  const int lane = tid & 31;
  const int hi   = lane >> 4;
  const int ln   = lane & 15;
  const int q0   = qb * 128 + wave * 16;

  // staging addresses (per thread, advance with kb0)
  const int kkey = tid >> 3;                 // 0..31
  const int kdb  = (tid & 7) * 16;           // 0..112
  const unsigned short* gK = k_bf + ((size_t)hkv * S_LEN + kkey) * HD + kdb;
  const int kofs = kkey * 136 + kdb;
  const int vd   = tid >> 1;                 // 0..127
  const int vh   = tid & 1;
  const unsigned short* gV = v_t + ((size_t)hkv * HD + vd) * S_LEN + vh * 16;
  const int vofs = vd * 40 + vh * 16;

  // Q fragments (16 rows x 128 d) straight from global
  FragAB qf[4];
  const unsigned short* qrow = q_bf + (size_t)(q0 + ln) * (NHQ * HD) + h * HD;
  #pragma unroll
  for (int i = 0; i < 4; ++i)
    #pragma unroll
    for (int p = 0; p < 8; ++p) {
      int kb = i * 32 + (p < 4 ? 2 * p : 2 * p + 8) + 8 * hi;
      qf[i].u[p] = *(const unsigned*)(qrow + kb);
    }

  v8f o[8] = {};
  float mrow[8], lrow[8];
  #pragma unroll
  for (int r = 0; r < 8; ++r) { mrow[r] = NEG_BIG; lrow[r] = 0.f; }
  const int myqmax = q0 + 15;
  const int kend   = qb * 128 + 128;

  // prologue: stage key block 0 into buffer 0
  async_ld_b128(&Ks[0][kofs],     gK);
  async_ld_b128(&Ks[0][kofs + 8], gK + 8);
  async_ld_b128(&Vs[0][vofs],     gV);
  async_ld_b128(&Vs[0][vofs + 8], gV + 8);

  for (int kb0 = 0; kb0 < kend; kb0 += 32) {
    const int cur = (kb0 >> 5) & 1;
    if (kb0 + 32 < kend) {
      const int nb = cur ^ 1;
      async_ld_b128(&Ks[nb][kofs],     gK + (size_t)(kb0 + 32) * HD);
      async_ld_b128(&Ks[nb][kofs + 8], gK + (size_t)(kb0 + 32) * HD + 8);
      async_ld_b128(&Vs[nb][vofs],     gV + kb0 + 32);
      async_ld_b128(&Vs[nb][vofs + 8], gV + kb0 + 40);
      wait_async4();
    } else {
      wait_async0();
    }
    __syncthreads();

    if (kb0 <= myqmax) {
      // scores: two 16x16 tiles (keys kb0..+15, kb0+16..+31)
      v8f sc[2] = {};
      #pragma unroll
      for (int t = 0; t < 2; ++t) {
        #pragma unroll
        for (int i = 0; i < 4; ++i) {
          FragAB kf;
          #pragma unroll
          for (int p = 0; p < 8; ++p)
            kf.u[p] = *(const unsigned*)(&Ks[cur][(t * 16 + ln) * 136 + i * 32 + (p < 4 ? 2 * p : 2 * p + 8) + 8 * hi]);
          sc[t] = __builtin_amdgcn_wmma_f32_16x16x32_bf16(
              false, qf[i].bf, false, kf.bf, (short)0, sc[t], false, false);
        }
      }
      // scale + causal mask (D layout: row = r + 8*hi, col key = kb0 + t*16 + ln)
      #pragma unroll
      for (int t = 0; t < 2; ++t)
        #pragma unroll
        for (int r = 0; r < 8; ++r) {
          int key = kb0 + t * 16 + ln;
          int qq  = q0 + r + 8 * hi;
          float v = sc[t][r] * ATT_SCALE;
          sc[t][r] = (key > qq) ? NEG_BIG : v;
        }
      // online softmax stats per row (reduce across 16-lane half groups)
      float alpha[8];
      #pragma unroll
      for (int r = 0; r < 8; ++r) {
        float mx = fmaxf(sc[0][r], sc[1][r]);
        #pragma unroll
        for (int m = 8; m >= 1; m >>= 1) mx = fmaxf(mx, __shfl_xor(mx, m, 32));
        float mnew = fmaxf(mrow[r], mx);
        float a  = __expf(mrow[r] - mnew);
        float p0 = __expf(sc[0][r] - mnew);
        float p1 = __expf(sc[1][r] - mnew);
        float srow = p0 + p1;
        #pragma unroll
        for (int m = 8; m >= 1; m >>= 1) srow += __shfl_xor(srow, m, 32);
        lrow[r] = lrow[r] * a + srow;
        mrow[r] = mnew;
        alpha[r] = a;
        sc[0][r] = p0; sc[1][r] = p1;
      }
      // rescale accumulators
      #pragma unroll
      for (int c = 0; c < 8; ++c)
        #pragma unroll
        for (int r = 0; r < 8; ++r) o[c][r] *= alpha[r];
      // transpose P (D layout -> A layout) through per-wave LDS
      unsigned short* ps = &Ps[wave][0];
      #pragma unroll
      for (int t = 0; t < 2; ++t)
        #pragma unroll
        for (int r = 0; r < 8; ++r)
          ps[(r + 8 * hi) * 40 + t * 16 + ln] = f2bf(sc[t][r]);
      asm volatile("s_wait_dscnt 0" ::: "memory");
      FragAB pa;
      #pragma unroll
      for (int p = 0; p < 8; ++p) {
        int kb = (p < 4 ? 2 * p : 2 * p + 8) + 8 * hi;
        pa.u[p] = *(const unsigned*)(ps + ln * 40 + kb);
      }
      // O += P @ V  (8 d-slices of 16)
      #pragma unroll
      for (int c = 0; c < 8; ++c) {
        FragAB vb;
        #pragma unroll
        for (int p = 0; p < 8; ++p)
          vb.u[p] = *(const unsigned*)(&Vs[cur][(c * 16 + ln) * 40 + (p < 4 ? 2 * p : 2 * p + 8) + 8 * hi]);
        o[c] = __builtin_amdgcn_wmma_f32_16x16x32_bf16(
            false, pa.bf, false, vb.bf, (short)0, o[c], false, false);
      }
    }
    __syncthreads();
  }

  float inv[8];
  #pragma unroll
  for (int r = 0; r < 8; ++r) inv[r] = 1.0f / lrow[r];
  #pragma unroll
  for (int c = 0; c < 8; ++c)
    #pragma unroll
    for (int r = 0; r < 8; ++r) {
      int rr = q0 + r + 8 * hi;
      o_bf[(size_t)rr * (NHQ * HD) + h * HD + c * 16 + ln] = f2bf(o[c][r] * inv[r]);
    }
}

// ---------------- host side ----------------
extern "C" void kernel_launch(void* const* d_in, const int* in_sizes, int n_in,
                              void* d_out, int out_size, void* d_ws, size_t ws_size,
                              hipStream_t stream) {
  (void)in_sizes; (void)n_in; (void)out_size; (void)ws_size;
  const float* hidden = (const float*)d_in[0];
  const float* cosb   = (const float*)d_in[1];
  const float* sinb   = (const float*)d_in[2];
  // d_in[3] = attention_mask (causal, implicit)
  const float* wq     = (const float*)d_in[4];
  const float* wk     = (const float*)d_in[5];
  const float* wv     = (const float*)d_in[6];
  const float* wo     = (const float*)d_in[7];
  const float* qnw    = (const float*)d_in[8];
  const float* knw    = (const float*)d_in[9];
  float* out          = (float*)d_out;

  unsigned char* ws = (unsigned char*)d_ws;
  size_t off = 0;
  auto take = [&](size_t bytes) { void* p = ws + off; off = (off + bytes + 255) & ~(size_t)255; return p; };
  unsigned short* hid_bf = (unsigned short*)take((size_t)S_LEN * HID * 2);        // 8 MB
  unsigned short* w_t    = (unsigned short*)take((size_t)HID * 4096 * 2);         // 16 MB  [4096 n][2048 k]
  float*          qkv    = (float*)take((size_t)S_LEN * 4096 * 4);                // 32 MB
  unsigned short* q_bf   = (unsigned short*)take((size_t)S_LEN * NHQ * HD * 2);   // 8 MB
  unsigned short* k_bf   = (unsigned short*)take((size_t)NKV * S_LEN * HD * 2);   // 4 MB
  unsigned short* v_t    = (unsigned short*)take((size_t)NKV * HD * S_LEN * 2);   // 4 MB
  unsigned short* o_bf   = (unsigned short*)take((size_t)S_LEN * NHQ * HD * 2);   // 8 MB
  unsigned short* wo_t   = (unsigned short*)take((size_t)HID * HID * 2);          // 8 MB [n][k]

  cvt_bf16<<<(S_LEN * HID) / 256, 256, 0, stream>>>(hidden, hid_bf, S_LEN * HID);
  pack_w_t<<<(HID * 4096) / 256, 256, 0, stream>>>(wq, wk, wv, w_t);
  cvt_bf16_t<<<(HID * HID) / 256, 256, 0, stream>>>(wo, wo_t);

  // QKV projection: [2048 x 2048] * [2048 x 4096] -> f32
  gemm_bf16<<<(4096 / 256) * (S_LEN / 128), 256, 0, stream>>>(hid_bf, w_t, qkv, S_LEN, 4096, HID);

  norm_rope<<<(S_LEN * 32 * 32) / 256, 256, 0, stream>>>(qkv, cosb, sinb, qnw, knw, q_bf, k_bf, v_t);

  attn<<<NHQ * (S_LEN / 128), 256, 0, stream>>>(q_bf, k_bf, v_t, o_bf);

  // output projection -> f32 d_out
  gemm_bf16<<<(HID / 256) * (S_LEN / 128), 256, 0, stream>>>(o_bf, wo_t, out, S_LEN, HID, HID);
}
